// MultiHeadAttention_6863357739514
// MI455X (gfx1250) — matrix operand order repaired and checked by reading, not compiled, not run on hardware
//
#include <hip/hip_runtime.h>
#include <math.h>
#include <stdint.h>
#include <stddef.h>

// ---------------------------------------------------------------------------
// MHA forward for MI455X (gfx1250), wave32, bf16 WMMA with fp32 accumulation.
// Data movement showcase: TDM (tensor_load_to_lds / TENSORcnt) for K tiles,
// global_load_async_to_lds (ASYNCcnt) for V tiles, software-pipelined GEMMs.
// ---------------------------------------------------------------------------

typedef __bf16 bf16_t;
typedef __attribute__((ext_vector_type(16))) __bf16 v16bf;
typedef __attribute__((ext_vector_type(8)))  __bf16 v8bf;
typedef __attribute__((ext_vector_type(4)))  __bf16 v4bf;
typedef __attribute__((ext_vector_type(8)))  float  v8f;
typedef __attribute__((ext_vector_type(4)))  uint32_t u32x4;
typedef __attribute__((ext_vector_type(8)))  uint32_t u32x8;

#define D_SIZE 1024
#define SEQ    2048
#define NB     2
#define NH     16
#define HD     64
#define NTOK   (NB * SEQ)      // 4096 tokens

union V16U { v16bf v; v8bf h[2]; };

// --- WMMA operand loaders (wave32 16-bit layouts per CDNA5 ISA 7.12.2) ------
// A matrix 16x32 (MxK), source row-major with leading dim lda:
//   lanes 0-15 : row M=lane,   K in {0..7, 16..23}
//   lanes 16-31: row M=lane-16,K in {8..15, 24..31}
__device__ inline v16bf load_tileA(const bf16_t* __restrict__ p, int lda,
                                   int row0, int k0) {
  const int lane = threadIdx.x & 31;
  const int half = lane >> 4;
  const int r    = lane & 15;
  const bf16_t* base = p + (size_t)(row0 + r) * lda + k0 + half * 8;
  V16U t;
  t.h[0] = *(const v8bf*)(base);        // K idx 0..7   (+half*8)
  t.h[1] = *(const v8bf*)(base + 16);   // K idx 16..23 (+half*8)
  return t.v;
}

// B matrix 32x16 (KxN). Source holds columns contiguously: row n of `p`
// (leading dim ldb) supplies B[:, n].  lanes 0-15: col n=lane, K in {0..7, 16..23};
// lanes 16-31: col n=lane-16, K in {8..15, 24..31}.
__device__ inline v16bf load_tileB(const bf16_t* __restrict__ p, int ldb,
                                   int n0, int k0) {
  const int lane = threadIdx.x & 31;
  const int n    = lane & 15;
  const int koff = (lane >> 4) << 3;
  const bf16_t* base = p + (size_t)(n0 + n) * ldb + k0 + koff;
  V16U t;
  t.h[0] = *(const v8bf*)(base);
  t.h[1] = *(const v8bf*)(base + 16);
  return t.v;
}

__device__ inline v8f wmma_bf16(v16bf a, v16bf b, v8f c) {
  return __builtin_amdgcn_wmma_f32_16x16x32_bf16(
      /*neg_a=*/false, a, /*neg_b=*/false, b,
      /*c_mod=*/(short)0, c, /*reuse_a=*/false, /*reuse_b=*/false);
}

// --- Tensor Data Mover: DMA one contiguous 32x64 bf16 tile (4KB) to LDS. ----
// D# per CDNA5 ISA 8.3/8.4: 2 SGPR groups (2D tensor), data_size=2B,
// tile 64 x 32, dim0 stride 64 elements. Tracked on TENSORcnt.
__device__ inline void tdm_load_K(const bf16_t* src, uint32_t ldsoff) {
  const uint64_t ga = (uint64_t)(uintptr_t)src;
  u32x4 g0;
  g0[0] = 1u;                                  // count=1, user mode
  g0[1] = ldsoff;                              // lds_addr  [63:32]
  g0[2] = (uint32_t)ga;                        // global_addr[31:0]
  g0[3] = (uint32_t)(ga >> 32) | (2u << 30);   // global_addr[56:32] | type=2
  u32x8 g1;
  g1[0] = 0x10000u;           // wg_mask=0, data_size=1 (2 bytes)
  g1[1] = 64u << 16;          // tensor_dim0 = 64  (bits 79:48 lo half)
  g1[2] = 32u << 16;          // tensor_dim1 = 32  (bits 111:80 lo half)
  g1[3] = 64u << 16;          // tile_dim0 = 64    (bits 127:112)
  g1[4] = 32u;                // tile_dim1 = 32, tile_dim2 = 0
  g1[5] = 64u;                // tensor_dim0_stride = 64 (elements)
  g1[6] = 0u;                 // stride hi / dim1_stride lo
  g1[7] = 0u;                 // dim1_stride hi
  asm volatile("tensor_load_to_lds %0, %1" :: "s"(g0), "s"(g1) : "memory");
}

// --- Async DMA of the 64x32 bf16 V panel (4KB, rows strided by SEQ) ---------
// 8 x (32 lanes x B128) on the ASYNCcnt pipe.
__device__ inline void stage_V_async(const bf16_t* __restrict__ src,
                                     bf16_t* dst) {
  const int lane = threadIdx.x & 31;
#pragma unroll
  for (int it = 0; it < 8; ++it) {
    const int c   = it * 32 + lane;            // 16-byte chunk index 0..255
    const int row = c >> 2;                    // d index 0..63
    const int qo  = (c & 3) * 8;               // element offset within row
    const uint64_t ga = (uint64_t)(uintptr_t)(src + (size_t)row * SEQ + qo);
    const uint32_t la = (uint32_t)(uintptr_t)(dst + row * 32 + qo);
    asm volatile("global_load_async_to_lds_b128 %0, %1, off"
                 :: "v"(la), "v"(ga) : "memory");
  }
}

// --------------------------- fp32 -> bf16 conversion ------------------------
__global__ void cvt_f32_bf16(const float4* __restrict__ s,
                             bf16_t* __restrict__ d, int n4) {
  int i = blockIdx.x * blockDim.x + threadIdx.x;
  if (i < n4) {
    float4 f = s[i];
    v4bf o;
    o[0] = (bf16_t)f.x; o[1] = (bf16_t)f.y;
    o[2] = (bf16_t)f.z; o[3] = (bf16_t)f.w;
    *(v4bf*)(d + (size_t)i * 4) = o;
  }
}

// --------------------------- fused QKV projection ---------------------------
// out = x @ W^T + b ; one wave computes a 16x64 output strip, k-loop software
// pipelined (next tiles in flight while current WMMAs issue).
// Q,K written [B,H,S,Hd]; V written transposed [B,H,Hd,S] for P.V B-tiles.
__global__ __launch_bounds__(256)
void qkv_proj(const bf16_t* __restrict__ xbf,
              const bf16_t* __restrict__ wq, const bf16_t* __restrict__ wk,
              const bf16_t* __restrict__ wv,
              const float* __restrict__ bq, const float* __restrict__ bk,
              const float* __restrict__ bv,
              bf16_t* __restrict__ Qo, bf16_t* __restrict__ Ko,
              bf16_t* __restrict__ Vt) {
  const int wave = blockIdx.x * (blockDim.x >> 5) + (threadIdx.x >> 5);
  const int m0   = (wave / 48) * 16;   // 256 m-tiles
  const int c    = wave % 48;          // 48 column strips (3 mats x 16)
  const int sel  = c >> 4;             // 0=Q 1=K 2=V (wave-uniform)
  const int n0   = (c & 15) * 64;

  const bf16_t* W    = (sel == 0) ? wq : (sel == 1) ? wk : wv;
  const float*  bias = (sel == 0) ? bq : (sel == 1) ? bk : bv;

  v8f acc[4] = {};
  v16bf a  = load_tileA(xbf, D_SIZE, m0, 0);
  v16bf b0 = load_tileB(W, D_SIZE, n0 +  0, 0);
  v16bf b1 = load_tileB(W, D_SIZE, n0 + 16, 0);
  v16bf b2 = load_tileB(W, D_SIZE, n0 + 32, 0);
  v16bf b3 = load_tileB(W, D_SIZE, n0 + 48, 0);
  for (int kk = 32; kk < D_SIZE; kk += 32) {
    __builtin_prefetch(xbf + (size_t)m0 * D_SIZE + kk + 32, 0, 1);
    __builtin_prefetch(W + (size_t)n0 * D_SIZE + kk + 32, 0, 1);
    v16bf an  = load_tileA(xbf, D_SIZE, m0, kk);
    v16bf bn0 = load_tileB(W, D_SIZE, n0 +  0, kk);
    v16bf bn1 = load_tileB(W, D_SIZE, n0 + 16, kk);
    v16bf bn2 = load_tileB(W, D_SIZE, n0 + 32, kk);
    v16bf bn3 = load_tileB(W, D_SIZE, n0 + 48, kk);
    acc[0] = wmma_bf16(a, b0, acc[0]);
    acc[1] = wmma_bf16(a, b1, acc[1]);
    acc[2] = wmma_bf16(a, b2, acc[2]);
    acc[3] = wmma_bf16(a, b3, acc[3]);
    a = an; b0 = bn0; b1 = bn1; b2 = bn2; b3 = bn3;
  }
  acc[0] = wmma_bf16(a, b0, acc[0]);
  acc[1] = wmma_bf16(a, b1, acc[1]);
  acc[2] = wmma_bf16(a, b2, acc[2]);
  acc[3] = wmma_bf16(a, b3, acc[3]);

  const int lane = threadIdx.x & 31;
  const int g = lane >> 4, col = lane & 15;
#pragma unroll
  for (int t = 0; t < 4; ++t) {
    const int n = n0 + t * 16 + col;
    const float bval = bias[n];
    const int h = n >> 6, dd = n & 63;
#pragma unroll
    for (int r = 0; r < 8; ++r) {
      const int m  = m0 + r + 8 * g;     // token index
      const int bb = m >> 11, ss = m & (SEQ - 1);
      const float v = acc[t][r] + bval;
      if (sel == 0)
        Qo[(((size_t)(bb * NH + h)) * SEQ + ss) * HD + dd] = (bf16_t)v;
      else if (sel == 1)
        Ko[(((size_t)(bb * NH + h)) * SEQ + ss) * HD + dd] = (bf16_t)v;
      else
        Vt[(((size_t)(bb * NH + h)) * HD + dd) * SEQ + ss] = (bf16_t)v;
    }
  }
}

// ------------------------------ flash attention -----------------------------
// One wave per (b,h,q-tile of 16 rows). Mask keeps k >= q (reference triu),
// so the k loop starts at this wave's diagonal tile. Online softmax, fp32
// running stats. K pair-tiles DMA'd by the TDM (TENSORcnt), V panels by
// async-to-LDS (ASYNCcnt); both double-buffered one iteration ahead.
__global__ __launch_bounds__(256)
void flash_attn(const bf16_t* __restrict__ Q, const bf16_t* __restrict__ K,
                const bf16_t* __restrict__ Vt, bf16_t* __restrict__ attn) {
  __shared__ __align__(128) bf16_t kbuf[8][2][2048];  // 8KB/wave K dbl-buffer
  __shared__ __align__(128) bf16_t vbuf[8][2][2048];  // 8KB/wave V dbl-buffer
  __shared__ __align__(16)  bf16_t pbuf[8][16 * 32];  // 1KB/wave P staging
  const int wid  = threadIdx.x >> 5;
  const int wave = blockIdx.x * 8 + wid;
  const int qt = wave & 127;          // 128 q-tiles per (b,h)
  const int bh = wave >> 7;
  const int b  = bh >> 4, h = bh & 15;
  const int q0 = qt * 16;

  const bf16_t* Qbh = Q  + (size_t)bh * SEQ * HD;
  const bf16_t* Kbh = K  + (size_t)bh * SEQ * HD;
  const bf16_t* Vbh = Vt + (size_t)bh * HD * SEQ;

  const int lane = threadIdx.x & 31;
  const int g = lane >> 4, col = lane & 15;

  const v16bf aq0 = load_tileA(Qbh, HD, q0, 0);
  const v16bf aq1 = load_tileA(Qbh, HD, q0, 32);

  float mrun[8], lrun[8];
  v8f o[4] = {};
#pragma unroll
  for (int r = 0; r < 8; ++r) { mrun[r] = -__builtin_inff(); lrun[r] = 0.f; }

  const float scale = 0.03125f;                      // 1/sqrt(1024)
  const float FMIN  = -3.4028234663852886e38f;       // np.finfo(f32).min

  const uint32_t koff0 = (uint32_t)(uintptr_t)&kbuf[wid][0][0];
  const uint32_t koff1 = (uint32_t)(uintptr_t)&kbuf[wid][1][0];

  const int j0 = qt & ~1;
  tdm_load_K(Kbh + (size_t)(16 * j0) * HD, koff0);
  stage_V_async(Vbh + 16 * j0, &vbuf[wid][0][0]);
  int cur = 0;

  for (int j = j0; j < SEQ / 16; j += 2) {           // pair of 16-col k-tiles
    asm volatile("s_wait_tensorcnt 0x0" ::: "memory");
    asm volatile("s_wait_asynccnt 0x0" ::: "memory");
    if (j + 2 < SEQ / 16) {                          // stage next pair ahead
      tdm_load_K(Kbh + (size_t)(16 * (j + 2)) * HD, cur ? koff0 : koff1);
      stage_V_async(Vbh + 16 * (j + 2), &vbuf[wid][cur ^ 1][0]);
    }

    // ---- scores S = (Q Kt) * scale from LDS-staged K ----
    const bf16_t* kb = &kbuf[wid][cur][0];           // 32x64 row-major
    v16bf bk0 = load_tileB(kb, HD,  0,  0);
    v16bf bk1 = load_tileB(kb, HD,  0, 32);
    v16bf bk2 = load_tileB(kb, HD, 16,  0);
    v16bf bk3 = load_tileB(kb, HD, 16, 32);
    v8f s0 = {}, s1 = {};
    s0 = wmma_bf16(aq0, bk0, s0);
    s0 = wmma_bf16(aq1, bk1, s0);
    s1 = wmma_bf16(aq0, bk2, s1);
    s1 = wmma_bf16(aq1, bk3, s1);

    float p0[8], p1[8], tmax[8], tsum[8];
#pragma unroll
    for (int r = 0; r < 8; ++r) {
      const int qg  = q0 + r + 8 * g;
      const int k0g = 16 * j + col;
      const int k1g = 16 * (j + 1) + col;
      float v0 = s0[r] * scale;
      float v1 = s1[r] * scale;
      if (k0g < qg) v0 = FMIN;
      if (k1g < qg) v1 = FMIN;
      p0[r] = v0; p1[r] = v1;
      tmax[r] = fmaxf(v0, v1);
    }
    // row max across the 16 lanes holding this row's columns
#pragma unroll
    for (int r = 0; r < 8; ++r)
      for (int off = 1; off < 16; off <<= 1)
        tmax[r] = fmaxf(tmax[r], __shfl_xor(tmax[r], off, 32));
    float alpha[8];
#pragma unroll
    for (int r = 0; r < 8; ++r) {
      const float mn = fmaxf(mrun[r], tmax[r]);
      alpha[r] = __expf(mrun[r] - mn);
      mrun[r]  = mn;
      p0[r] = __expf(p0[r] - mn);
      p1[r] = __expf(p1[r] - mn);
      tsum[r] = p0[r] + p1[r];
    }
#pragma unroll
    for (int r = 0; r < 8; ++r)
      for (int off = 1; off < 16; off <<= 1)
        tsum[r] += __shfl_xor(tsum[r], off, 32);
#pragma unroll
    for (int r = 0; r < 8; ++r)
      lrun[r] = lrun[r] * alpha[r] + tsum[r];
    // rescale running output (per-row factors already uniform per lane group)
#pragma unroll
    for (int t = 0; t < 4; ++t)
#pragma unroll
      for (int r = 0; r < 8; ++r)
        o[t][r] *= alpha[r];

    // ---- P: C layout -> A layout via wave-private LDS ----
#pragma unroll
    for (int r = 0; r < 8; ++r) {
      const int row = r + 8 * g;
      pbuf[wid][row * 32 + col]      = (bf16_t)p0[r];
      pbuf[wid][row * 32 + 16 + col] = (bf16_t)p1[r];
    }
    asm volatile("s_wait_dscnt 0x0" ::: "memory");
    v16bf pa;
    {
      const int row  = lane & 15;
      const int half = lane >> 4;
      V16U t;
      t.h[0] = *(const v8bf*)&pbuf[wid][row * 32 + half * 8];
      t.h[1] = *(const v8bf*)&pbuf[wid][row * 32 + 16 + half * 8];
      pa = t.v;
    }
    // ---- O += P @ V from LDS-staged V (64 rows x 32 cols, ld=32) ----
    const bf16_t* vb = &vbuf[wid][cur][0];
    v16bf bv0 = load_tileB(vb, 32,  0, 0);
    v16bf bv1 = load_tileB(vb, 32, 16, 0);
    v16bf bv2 = load_tileB(vb, 32, 32, 0);
    v16bf bv3 = load_tileB(vb, 32, 48, 0);
    o[0] = wmma_bf16(pa, bv0, o[0]);
    o[1] = wmma_bf16(pa, bv1, o[1]);
    o[2] = wmma_bf16(pa, bv2, o[2]);
    o[3] = wmma_bf16(pa, bv3, o[3]);
    cur ^= 1;
  }

  // epilogue: divide by row sum, write token-major bf16 [B*S, D]
#pragma unroll
  for (int t = 0; t < 4; ++t) {
#pragma unroll
    for (int r = 0; r < 8; ++r) {
      const int qg = q0 + r + 8 * g;
      const float val = o[t][r] / lrun[r];
      attn[((size_t)(b * SEQ + qg)) * D_SIZE + h * HD + t * 16 + col] =
          (bf16_t)val;
    }
  }
}

// ------------------------------ output projection ---------------------------
__global__ __launch_bounds__(256)
void out_proj(const bf16_t* __restrict__ attn, const bf16_t* __restrict__ wo,
              const float* __restrict__ bo, float* __restrict__ out) {
  const int wave = blockIdx.x * 8 + (threadIdx.x >> 5);
  const int m0 = (wave >> 4) * 16;
  const int n0 = (wave & 15) * 64;

  v8f acc[4] = {};
  v16bf a  = load_tileA(attn, D_SIZE, m0, 0);
  v16bf b0 = load_tileB(wo, D_SIZE, n0 +  0, 0);
  v16bf b1 = load_tileB(wo, D_SIZE, n0 + 16, 0);
  v16bf b2 = load_tileB(wo, D_SIZE, n0 + 32, 0);
  v16bf b3 = load_tileB(wo, D_SIZE, n0 + 48, 0);
  for (int kk = 32; kk < D_SIZE; kk += 32) {
    __builtin_prefetch(attn + (size_t)m0 * D_SIZE + kk + 32, 0, 1);
    __builtin_prefetch(wo + (size_t)n0 * D_SIZE + kk + 32, 0, 1);
    v16bf an  = load_tileA(attn, D_SIZE, m0, kk);
    v16bf bn0 = load_tileB(wo, D_SIZE, n0 +  0, kk);
    v16bf bn1 = load_tileB(wo, D_SIZE, n0 + 16, kk);
    v16bf bn2 = load_tileB(wo, D_SIZE, n0 + 32, kk);
    v16bf bn3 = load_tileB(wo, D_SIZE, n0 + 48, kk);
    acc[0] = wmma_bf16(a, b0, acc[0]);
    acc[1] = wmma_bf16(a, b1, acc[1]);
    acc[2] = wmma_bf16(a, b2, acc[2]);
    acc[3] = wmma_bf16(a, b3, acc[3]);
    a = an; b0 = bn0; b1 = bn1; b2 = bn2; b3 = bn3;
  }
  acc[0] = wmma_bf16(a, b0, acc[0]);
  acc[1] = wmma_bf16(a, b1, acc[1]);
  acc[2] = wmma_bf16(a, b2, acc[2]);
  acc[3] = wmma_bf16(a, b3, acc[3]);

  const int lane = threadIdx.x & 31;
  const int g = lane >> 4, col = lane & 15;
#pragma unroll
  for (int t = 0; t < 4; ++t) {
    const int n = n0 + t * 16 + col;
    const float bval = bo[n];
#pragma unroll
    for (int r = 0; r < 8; ++r) {
      const int m = m0 + r + 8 * g;
      out[(size_t)m * D_SIZE + n] = acc[t][r] + bval;
    }
  }
}

// --------------------------------- launcher ---------------------------------
extern "C" void kernel_launch(void* const* d_in, const int* in_sizes, int n_in,
                              void* d_out, int out_size, void* d_ws,
                              size_t ws_size, hipStream_t stream) {
  (void)in_sizes; (void)n_in; (void)out_size; (void)ws_size;

  const float* x  = (const float*)d_in[0];
  const float* Wq = (const float*)d_in[1];
  const float* bq = (const float*)d_in[2];
  const float* Wk = (const float*)d_in[3];
  const float* bk = (const float*)d_in[4];
  const float* Wv = (const float*)d_in[5];
  const float* bv = (const float*)d_in[6];
  const float* Wo = (const float*)d_in[7];
  const float* bo = (const float*)d_in[8];
  float* out = (float*)d_out;

  const size_t NX = (size_t)NTOK * D_SIZE;       // 4,194,304
  const size_t NW = (size_t)D_SIZE * D_SIZE;     // 1,048,576

  bf16_t* ws = (bf16_t*)d_ws;
  size_t off = 0;
  bf16_t* xbf   = ws + off; off += NX;   // x in bf16
  bf16_t* wqb   = ws + off; off += NW;
  bf16_t* wkb   = ws + off; off += NW;
  bf16_t* wvb   = ws + off; off += NW;
  bf16_t* wob   = ws + off; off += NW;
  bf16_t* Qb    = ws + off; off += NX;   // [B,H,S,Hd]
  bf16_t* Kb    = ws + off; off += NX;   // [B,H,S,Hd]
  bf16_t* Vtb   = ws + off; off += NX;   // [B,H,Hd,S]
  bf16_t* attnb = ws + off; off += NX;   // [B*S, D]
  // total: ~50.3 MB of workspace

  const int T = 256;
  cvt_f32_bf16<<<(int)((NX / 4 + T - 1) / T), T, 0, stream>>>(
      (const float4*)x, xbf, (int)(NX / 4));
  cvt_f32_bf16<<<(int)((NW / 4 + T - 1) / T), T, 0, stream>>>(
      (const float4*)Wq, wqb, (int)(NW / 4));
  cvt_f32_bf16<<<(int)((NW / 4 + T - 1) / T), T, 0, stream>>>(
      (const float4*)Wk, wkb, (int)(NW / 4));
  cvt_f32_bf16<<<(int)((NW / 4 + T - 1) / T), T, 0, stream>>>(
      (const float4*)Wv, wvb, (int)(NW / 4));
  cvt_f32_bf16<<<(int)((NW / 4 + T - 1) / T), T, 0, stream>>>(
      (const float4*)Wo, wob, (int)(NW / 4));

  // 256 m-tiles * 48 strips = 12288 waves / 8 waves-per-block
  qkv_proj<<<1536, 256, 0, stream>>>(xbf, wqb, wkb, wvb, bq, bk, bv,
                                     Qb, Kb, Vtb);
  // 2*16*128 = 4096 q-tile waves / 8 per block
  flash_attn<<<512, 256, 0, stream>>>(Qb, Kb, Vtb, attnb);
  // 256 m-tiles * 16 strips = 4096 waves / 8 per block
  out_proj<<<512, 256, 0, stream>>>(attnb, wob, bo, out);
}
